// MooreGRU_33870112096356
// MI455X (gfx1250) — matrix operand order repaired and checked by reading, not compiled, not run on hardware
//
#include <hip/hip_runtime.h>
#include <hip/hip_bf16.h>

// Model dims (compile-time constants from reference)
#define BB   128
#define TT   512
#define DD   256
#define G3   768          // 3*D
#define NS   64           // num states
#define MTOT (BB*TT)      // 65536 rows

typedef __attribute__((ext_vector_type(16))) _Float16 v16h;
typedef __attribute__((ext_vector_type(8)))  _Float16 v8h;
typedef __attribute__((ext_vector_type(8)))  float    v8f;

// ---------------------------------------------------------------------------
// CDNA5 async global->LDS copy (ASYNCcnt path) via inline asm (portable
// across toolchains whose builtin prototypes differ). GV addressing mode:
//   global_load_async_to_lds_b128 vDst(LDS byte addr), vAddr(64b global), off
// A generic pointer to __shared__ memory carries the LDS byte offset in its
// low 32 bits (flat-LDS aperture mapping), so integer truncation yields VDST.
// ---------------------------------------------------------------------------
#if defined(__gfx1250__)
#define ASYNC_LDS 1
__device__ __forceinline__ void async_copy16(void* lds, const void* g) {
  unsigned loff = (unsigned)(uintptr_t)lds;
  unsigned long long ga = (unsigned long long)(uintptr_t)g;
  asm volatile("global_load_async_to_lds_b128 %0, %1, off"
               :: "v"(loff), "v"(ga) : "memory");
}
#define WAIT_ASYNC(n) asm volatile("s_wait_asynccnt " #n ::: "memory")
#else
#define ASYNC_LDS 0
#define WAIT_ASYNC(n)
#endif

__device__ __forceinline__ v8f wmma_f16(v16h a, v16h b, v8f c) {
  // D = A(16x32 f16) * B(32x16 f16) + C(16x16 f32)
  return __builtin_amdgcn_wmma_f32_16x16x32_f16(
      /*neg_a=*/false, a, /*neg_b=*/false, b,
      /*c_mod=*/(short)0, c, /*reuse_a=*/false, /*reuse_b=*/false);
}

// A fragment (16x32, f16). Source: row-major [16][ld] (ld in elements).
// ISA 7.12.2: lane (r=l&15, hf=l>>4) row M=r; elems 0..7 -> K=hf*8+e,
// elems 8..15 -> K=16+hf*8+(e-8). Two contiguous 16B chunks per lane.
__device__ __forceinline__ v16h load_A(const _Float16* base, int ld, int lane) {
  const int r = lane & 15, hf = lane >> 4;
  const _Float16* p = base + r * ld + hf * 8;
  v8h lo = *(const v8h*)(p);
  v8h hi = *(const v8h*)(p + 16);
  v16h a;
#pragma unroll
  for (int i = 0; i < 8; ++i) { a[i] = lo[i]; a[i + 8] = hi[i]; }
  return a;
}

// B fragment (32x16, f16). Source: row-major [N][K] (weight [g][d]), stride ld.
// Lane holds column N=r, elems 0..7 -> K=hf*8+e, elems 8..15 -> K=16+hf*8+(e-8).
__device__ __forceinline__ v16h load_B(const _Float16* base, int ld, int lane) {
  const int r = lane & 15, hf = lane >> 4;
  const _Float16* p = base + r * ld + hf * 8;
  v8h lo = *(const v8h*)(p);
  v8h hi = *(const v8h*)(p + 16);
  v16h b;
#pragma unroll
  for (int i = 0; i < 8; ++i) { b[i] = lo[i]; b[i + 8] = hi[i]; }
  return b;
}

__device__ __forceinline__ float sigmoidf_(float x) {
  return 1.0f / (1.0f + __expf(-x));
}

// ---------------------------------------------------------------------------
// 1) Embedding gather -> f16 activations  x[m, d] = emb[ids[m], d]
// ---------------------------------------------------------------------------
__global__ __launch_bounds__(256) void k_embed(const int* __restrict__ ids,
                                               const float* __restrict__ emb,
                                               _Float16* __restrict__ xo) {
  long i = (long)blockIdx.x * 256 + threadIdx.x;  // over MTOT*DD
  long m = i >> 8;          // /256
  int  d = (int)(i & 255);
  xo[i] = (_Float16)emb[(long)ids[m] * DD + d];
}

// ---------------------------------------------------------------------------
// 2) Convert weights f32 -> f16, zero loss accumulators
// ---------------------------------------------------------------------------
__global__ __launch_bounds__(256) void k_cvt(const float* __restrict__ wih,
                                             const float* __restrict__ whh,
                                             const float* __restrict__ hw,
                                             _Float16* __restrict__ wih16,
                                             _Float16* __restrict__ whh16,
                                             _Float16* __restrict__ hw16,
                                             float* __restrict__ scal) {
  long i = (long)blockIdx.x * 256 + threadIdx.x;
  const long NW = 2L * G3 * DD;  // both layers
  if (i < NW) { wih16[i] = (_Float16)wih[i]; whh16[i] = (_Float16)whh[i]; }
  if (i < NS * DD) hw16[i] = (_Float16)hw[i];
  if (i < 2) scal[i] = 0.0f;
}

// ---------------------------------------------------------------------------
// 3) Input projection GEMM: xg[t, b, g] = x[m=b*T+t, :] . w_ih[g, :] + b_ih[g]
//    One block per 16-row M-tile, 8 waves x 6 N-tiles. A staged in LDS
//    (async global->LDS when available).
// ---------------------------------------------------------------------------
__global__ __launch_bounds__(256) void k_xg_gemm(const _Float16* __restrict__ x,
                                                 const _Float16* __restrict__ w,
                                                 const float* __restrict__ bias,
                                                 float* __restrict__ xg) {
  __shared__ _Float16 sA[16 * DD];  // 8 KB
  const int tid = threadIdx.x, lane = tid & 31, wave = tid >> 5;
  const int r = lane & 15, hf = lane >> 4;
  const long m0 = (long)blockIdx.x * 16;

  // Stage A tile (16x256 f16 = 8KB = 512 x 16B) into LDS, 2 chunks/thread.
  {
    const uint4* gsrc = (const uint4*)(x + m0 * DD);
    uint4* sdst = (uint4*)sA;
#if ASYNC_LDS
    async_copy16(&sdst[tid],       &gsrc[tid]);
    async_copy16(&sdst[tid + 256], &gsrc[tid + 256]);
    WAIT_ASYNC(0);
#else
    sdst[tid]       = gsrc[tid];
    sdst[tid + 256] = gsrc[tid + 256];
#endif
  }
  __syncthreads();

  v16h afrag[8];
#pragma unroll
  for (int k = 0; k < 8; ++k) afrag[k] = load_A(sA + k * 32, DD, lane);

  // Each wave: N-tiles j = wave*6 .. wave*6+5 (48 total)
  for (int jj = 0; jj < 6; ++jj) {
    const int j = wave * 6 + jj;
    v8f acc;
    const float bv = bias[j * 16 + r];
#pragma unroll
    for (int v = 0; v < 8; ++v) acc[v] = bv;
#pragma unroll
    for (int k = 0; k < 8; ++k) {
      v16h bf = load_B(w + (long)(j * 16) * DD + k * 32, DD, lane);
      acc = wmma_f16(afrag[k], bf, acc);
    }
    // scatter-store into [T, B, 768] (scan-friendly layout)
#pragma unroll
    for (int v = 0; v < 8; ++v) {
      const long m = m0 + v + 8 * hf;
      const long b = m >> 9;          // /T (T=512)
      const long t = m & (TT - 1);
      xg[(t * BB + b) * G3 + j * 16 + r] = acc[v];
    }
  }
}

// ---------------------------------------------------------------------------
// 4) GRU recurrent scan. One block per 16-batch-row tile (8 blocks).
//    h resident in LDS (f16 copy -> WMMA A-fragments, f32 master for update).
//    Per step: hg[16,768] = h . w_hh^T + b_hh via 48x8 WMMAs over 8 waves.
//    xg slab (48KB/step) double-buffered into LDS with async copies so the
//    gate math never touches global memory on the critical path.
// ---------------------------------------------------------------------------
__global__ __launch_bounds__(256) void k_gru_scan(const float* __restrict__ xg,   // [T,B,768]
                                                  const _Float16* __restrict__ whh,// [768,256] f16
                                                  const float* __restrict__ bhh,   // [768]
                                                  _Float16* __restrict__ xout) {   // [B,T,256] f16
  __shared__ _Float16 h16[16 * DD];  // 8 KB
  __shared__ float    h32[16 * DD];  // 16 KB
#if ASYNC_LDS
  __shared__ float    sxg[2][16 * G3];  // 2 x 48 KB double buffer
#endif
  const int tid = threadIdx.x, lane = tid & 31, wave = tid >> 5;
  const int r = lane & 15, hf = lane >> 4;
  const long b0 = (long)blockIdx.x * 16;

  for (int i = tid; i < 16 * DD; i += 256) { h16[i] = (_Float16)0.0f; h32[i] = 0.0f; }

  // recurrent biases for this wave's two d-tiles x 3 gates
  float bias_v[2][3];
#pragma unroll
  for (int jj = 0; jj < 2; ++jj)
#pragma unroll
    for (int g = 0; g < 3; ++g)
      bias_v[jj][g] = bhh[g * DD + (2 * wave + jj) * 16 + r];

#if ASYNC_LDS
  // Prime the pipeline: slab for t=0 (12 x 16B per thread).
  {
    const float* src = xg + b0 * G3;
#pragma unroll
    for (int it = 0; it < 12; ++it) {
      const int f = (it * 256 + tid) * 4;
      async_copy16(&sxg[0][f], src + f);
    }
  }
#endif
  __syncthreads();

  for (int t = 0; t < TT; ++t) {
    // Read old h as A fragments (all waves read full K range).
    v16h afrag[8];
#pragma unroll
    for (int k = 0; k < 8; ++k) afrag[k] = load_A(h16 + k * 32, DD, lane);

#if ASYNC_LDS
    // Prefetch next step's xg slab while this step's WMMAs run.
    if (t + 1 < TT) {
      const float* src = xg + ((long)(t + 1) * BB + b0) * G3;
#pragma unroll
      for (int it = 0; it < 12; ++it) {
        const int f = (it * 256 + tid) * 4;
        async_copy16(&sxg[(t + 1) & 1][f], src + f);
      }
      WAIT_ASYNC(12);  // async loads complete in order: slab t has landed
    } else {
      WAIT_ASYNC(0);
    }
    const float* xgt = &sxg[t & 1][0];
#else
    const float* xgt = xg + ((long)t * BB + b0) * G3;
    if (t + 1 < TT)
      __builtin_prefetch(xgt + (long)BB * G3 + tid * 32, 0, 1);
#endif
    __syncthreads();  // h16 reads done by all waves; slab t visible to all

    for (int jj = 0; jj < 2; ++jj) {
      const int j = 2 * wave + jj;  // d-tile 0..15
      v8f acc[3];
#pragma unroll
      for (int g = 0; g < 3; ++g)
#pragma unroll
        for (int v = 0; v < 8; ++v) acc[g][v] = bias_v[jj][g];

#pragma unroll
      for (int k = 0; k < 8; ++k) {
#pragma unroll
        for (int g = 0; g < 3; ++g) {
          v16h bf = load_B(whh + (long)(g * DD + j * 16) * DD + k * 32, DD, lane);
          acc[g] = wmma_f16(afrag[k], bf, acc[g]);
        }
      }

      const int d = j * 16 + r;
#pragma unroll
      for (int v = 0; v < 8; ++v) {
        const int m = v + 8 * hf;
        const float* xrow = xgt + (long)m * G3;
        const float xr = xrow[d], xz = xrow[DD + d], xn = xrow[2 * DD + d];
        const float rg = sigmoidf_(xr + acc[0][v]);
        const float zg = sigmoidf_(xz + acc[1][v]);
        const float ng = tanhf(xn + rg * acc[2][v]);
        const float hold = h32[m * DD + d];
        const float hnew = (1.0f - zg) * ng + zg * hold;
        h32[m * DD + d] = hnew;
        h16[m * DD + d] = (_Float16)hnew;
        xout[((b0 + m) * TT + t) * DD + d] = (_Float16)hnew;
      }
    }
    __syncthreads();  // h writes + slab reads complete before next iteration
  }
}

// ---------------------------------------------------------------------------
// 5) Head GEMM + masked cross-entropy. One wave per 16-row M-tile.
// ---------------------------------------------------------------------------
__global__ __launch_bounds__(32) void k_head(const _Float16* __restrict__ x,    // [M,256]
                                             const _Float16* __restrict__ hw,   // [64,256]
                                             const int* __restrict__ targets,   // [M]
                                             const unsigned char* __restrict__ mask, // [M] bool8
                                             float* __restrict__ logits,        // [M,64]
                                             float* __restrict__ scal) {        // {nll_sum, count}
  __shared__ float sl[16 * NS];  // 4 KB
  const int lane = threadIdx.x & 31;
  const int r = lane & 15, hf = lane >> 4;
  const long m0 = (long)blockIdx.x * 16;

  v16h afrag[8];
#pragma unroll
  for (int k = 0; k < 8; ++k) afrag[k] = load_A(x + m0 * DD + k * 32, DD, lane);

#pragma unroll
  for (int jn = 0; jn < 4; ++jn) {
    v8f acc;
#pragma unroll
    for (int v = 0; v < 8; ++v) acc[v] = 0.0f;
#pragma unroll
    for (int k = 0; k < 8; ++k) {
      v16h bf = load_B(hw + (long)(jn * 16) * DD + k * 32, DD, lane);
      acc = wmma_f16(afrag[k], bf, acc);
    }
#pragma unroll
    for (int v = 0; v < 8; ++v) {
      const int mloc = v + 8 * hf;
      const int n = jn * 16 + r;
      sl[mloc * NS + n] = acc[v];
      logits[(m0 + mloc) * NS + n] = acc[v];
    }
  }
  __syncthreads();

  if (lane < 16) {
    const long m = m0 + lane;
    float mx = -3.402823466e38f;
#pragma unroll
    for (int n = 0; n < NS; ++n) mx = fmaxf(mx, sl[lane * NS + n]);
    float s = 0.0f;
#pragma unroll
    for (int n = 0; n < NS; ++n) s += __expf(sl[lane * NS + n] - mx);
    const float lse = __logf(s) + mx;
    const int tgt = targets[m];
    const float nll = lse - sl[lane * NS + tgt];
    if (mask[m]) {
      atomicAdd(&scal[0], nll);
      atomicAdd(&scal[1], 1.0f);
    }
  }
}

__global__ void k_loss(const float* __restrict__ scal, float* __restrict__ out_loss) {
  const float s = scal[0], c = scal[1];
  out_loss[0] = (c > 0.0f) ? (s / fmaxf(c, 1.0f)) : 0.0f;
}

// ---------------------------------------------------------------------------
// Launch. Workspace requirement ~270 MB:
//   xA f16 32MB | xB f16 32MB | xg f32 192MB | wih16 768KB | whh16 768KB
//   | hw16 32KB | scalars
// ---------------------------------------------------------------------------
extern "C" void kernel_launch(void* const* d_in, const int* in_sizes, int n_in,
                              void* d_out, int out_size, void* d_ws, size_t ws_size,
                              hipStream_t stream) {
  const int*   ids    = (const int*)d_in[0];
  const int*   tgts   = (const int*)d_in[1];
  const unsigned char* msk = (const unsigned char*)d_in[2];
  const float* emb    = (const float*)d_in[3];
  const float* w_ih   = (const float*)d_in[4];   // [2,768,256]
  const float* w_hh   = (const float*)d_in[5];   // [2,768,256]
  const float* b_ih   = (const float*)d_in[6];   // [2,768]
  const float* b_hh   = (const float*)d_in[7];   // [2,768]
  const float* head_w = (const float*)d_in[8];   // [64,256]

  float* out_logits = (float*)d_out;                     // [B,T,64]
  float* out_loss   = (float*)d_out + (long)MTOT * NS;   // scalar

  char* p = (char*)d_ws;
  _Float16* xA    = (_Float16*)p; p += (size_t)MTOT * DD * 2;   // 32 MB
  _Float16* xB    = (_Float16*)p; p += (size_t)MTOT * DD * 2;   // 32 MB
  float*    xg    = (float*)p;    p += (size_t)MTOT * G3 * 4;   // 192 MB
  _Float16* wih16 = (_Float16*)p; p += (size_t)2 * G3 * DD * 2;
  _Float16* whh16 = (_Float16*)p; p += (size_t)2 * G3 * DD * 2;
  _Float16* hw16  = (_Float16*)p; p += (size_t)NS * DD * 2;
  float*    scal  = (float*)p;

  // 1) embedding -> f16
  k_embed<<<(MTOT * DD) / 256, 256, 0, stream>>>(ids, emb, xA);
  // 2) weight conversion + zero accumulators
  k_cvt<<<(2 * G3 * DD + 255) / 256, 256, 0, stream>>>(w_ih, w_hh, head_w,
                                                       wih16, whh16, hw16, scal);

  // layers
  _Float16* xin = xA;
  _Float16* xout = xB;
  for (int l = 0; l < 2; ++l) {
    k_xg_gemm<<<MTOT / 16, 256, 0, stream>>>(xin, wih16 + (size_t)l * G3 * DD,
                                             b_ih + (size_t)l * G3, xg);
    k_gru_scan<<<BB / 16, 256, 0, stream>>>(xg, whh16 + (size_t)l * G3 * DD,
                                            b_hh + (size_t)l * G3, xout);
    _Float16* tmp = xin; xin = xout; xout = tmp;
  }

  // head + loss (xin now holds layer-2 output)
  k_head<<<MTOT / 16, 32, 0, stream>>>(xin, hw16, tgts, msk, out_logits, scal);
  k_loss<<<1, 1, 0, stream>>>(scal, out_loss);
}
